// PointInt_49555332661490
// MI455X (gfx1250) — matrix operand order repaired and checked by reading, not compiled, not run on hardware
//
#include <hip/hip_runtime.h>

// ---------------------------------------------------------------------------
// KPConv depthwise kernel for MI455X (gfx1250, wave32).
// Per query n: w[h,k] = relu(1 - |p_h - kp_k| / 1.2)   (h<32, k<15)
//              weighted[k,c] = sum_h w[h,k] * x[ind[n,h], c]
//              out[n,c] = sum_k weighted[k,c] * dw[k,c] + bias[c]
// The [16x32]x[32x16] contraction maps exactly onto V_WMMA_F32_16X16X32_F16
// (H = 32 = WMMA K).  C=128 -> 8 N-tiles -> 8 WMMAs per query (one wave).
//
// Roofline: 6.1 GFLOP matmul; HBM traffic ~66 MB (x table 25.6 MB << 192 MB
// L2, so the 819 MB gather hits L2).  L2-gather/compute bound -> stage each
// query's 32 feature rows through LDS once (f32->f16), feed the matrix pipes.
// ---------------------------------------------------------------------------

typedef __attribute__((ext_vector_type(16))) _Float16    v16h;
typedef __attribute__((ext_vector_type(2)))  __fp16      v2fp16;   // cvt_pkrtz result type
typedef __attribute__((ext_vector_type(8)))  float       v8f;
typedef __attribute__((ext_vector_type(8)))  unsigned int v8u;

#define NQ   50000
#define HN   32      // neighbors
#define KP   15      // kernel points
#define CH   128     // channels
#define INV_EXTENT (1.0f / 1.2f)

#define WAVES_PER_BLOCK 4
#define THREADS (WAVES_PER_BLOCK * 32)
#define XT_STRIDE 34   // halves per channel row (32 used + 2 pad, odd word count)
#define DWT_STRIDE 20  // floats per dwT row (16 used + 4 pad; 20c mod 64 distinct)

__global__ __launch_bounds__(THREADS)
void kpconv_wmma_kernel(const float* __restrict__ q_pts,       // [N,3]
                        const float* __restrict__ s_pts,       // [M,3]
                        const long long* __restrict__ inds,    // [N,32] int64
                        const float* __restrict__ x,           // [N,128]
                        const float* __restrict__ kp,          // [15,3]
                        const float* __restrict__ dw,          // [15,128]
                        const float* __restrict__ bias,        // [128]
                        float* __restrict__ out)               // [N,128]
{
    // ---- LDS ------------------------------------------------------------
    __shared__ float     dwT[CH * DWT_STRIDE];                 // 10240 B (transposed, k-padded)
    __shared__ float     P[WAVES_PER_BLOCK][HN][4];            // 2048 B
    __shared__ long long IndS[WAVES_PER_BLOCK][HN];            // 1024 B
    __shared__ _Float16  XT[WAVES_PER_BLOCK][CH * XT_STRIDE];  // 34816 B
    // total ~47 KB -> ~6 blocks (24 waves) resident per 320 KB WGP

    const int tid  = threadIdx.x;
    const int wave = tid >> 5;
    const int lane = tid & 31;
    const int n    = blockIdx.x * WAVES_PER_BLOCK + wave;      // 50000 = 12500*4

    // ---- block-wide: preload dw TRANSPOSED: dwT[c][k], k padded to 16 ---
    // Coalesced reads dw[k*128 + c] (c = tid); k=15 slot zeroed so the
    // per-tile depthwise reduction needs no bounds select.
    {
        const int c = tid;                                     // THREADS == CH
        #pragma unroll
        for (int k = 0; k < 16; ++k)
            dwT[c * DWT_STRIDE + k] = (k < KP) ? dw[k * CH + c] : 0.0f;
    }

    // ---- stage 1: per-lane neighbor offset + index into LDS -------------
    const long long ind = inds[(long long)n * HN + lane];
    IndS[wave][lane] = ind;
    {
        const float qx = q_pts[n * 3 + 0];
        const float qy = q_pts[n * 3 + 1];
        const float qz = q_pts[n * 3 + 2];
        P[wave][lane][0] = s_pts[ind * 3 + 0] - qx;
        P[wave][lane][1] = s_pts[ind * 3 + 1] - qy;
        P[wave][lane][2] = s_pts[ind * 3 + 2] - qz;
    }
    // Only block-wide dependency is dwT; P/IndS/XT are wave-private, but
    // this single barrier covers the intra-wave P/IndS visibility too.
    __syncthreads();

    // ---- stage 2a: cooperative gather of 32 feature rows -> f16 LDS -----
    // Rows in pairs: one v_cvt_pk_rt packs the {h, h+1} dword a B-operand
    // lane will later load with a single ds_load_b32.
    #pragma unroll
    for (int hh = 0; hh < HN / 2; ++hh) {
        const long long i0 = IndS[wave][2 * hh + 0];
        const long long i1 = IndS[wave][2 * hh + 1];
        const float4 r0 = ((const float4*)(x + i0 * CH))[lane];   // c = 4*lane..
        const float4 r1 = ((const float4*)(x + i1 * CH))[lane];
        const float e0[4] = {r0.x, r0.y, r0.z, r0.w};
        const float e1[4] = {r1.x, r1.y, r1.z, r1.w};
        #pragma unroll
        for (int e = 0; e < 4; ++e) {
            // low half = row 2hh, high half = row 2hh+1 (one packed cvt)
            const v2fp16 pk = __builtin_amdgcn_cvt_pkrtz(e0[e], e1[e]);
            const unsigned int u = __builtin_bit_cast(unsigned int, pk);
            const int c = 4 * lane + e;
            *(unsigned int*)&XT[wave][c * XT_STRIDE + 2 * hh] = u;  // 4B aligned
        }
    }

    // ---- stage 2b: build A (w^T, 16x32 f16) directly in WMMA A-layout ---
    // Lane (l&15) = row M (kernel point); lanes 0-15 cover K-dim 0..7 & 16..23,
    // lanes 16-31 cover 8..15 & 24..31 (ISA 16-bit A 16x32 layout).
    const int M      = lane & 15;
    const int hiHalf = lane >> 4;
    const int kclamp = (M < KP) ? M : 0;
    const float kx = kp[kclamp * 3 + 0];
    const float ky = kp[kclamp * 3 + 1];
    const float kz = kp[kclamp * 3 + 2];

    v16h a;
    #pragma unroll
    for (int v = 0; v < 8; ++v) {
        const int hb = ((v & 4) ? 16 : 0) + 2 * (v & 3) + (hiHalf ? 8 : 0);
        #pragma unroll
        for (int e = 0; e < 2; ++e) {
            const int h = hb + e;
            const float dx = P[wave][h][0] - kx;
            const float dy = P[wave][h][1] - ky;
            const float dz = P[wave][h][2] - kz;
            const float d  = __builtin_sqrtf(dx * dx + dy * dy + dz * dz);
            float w = 1.0f - d * INV_EXTENT;
            w = (w > 0.0f) ? w : 0.0f;
            w = (M < KP) ? w : 0.0f;            // pad row 15 of A with zeros
            a[2 * v + e] = (_Float16)w;
        }
    }

    // Intra-wave ordering only: XT stores above vs. ds loads below are from
    // the SAME wave (LDS is in-order per wave, DScnt).  No block barrier --
    // keeps the 4 waves' gather->WMMA pipelines fully decoupled.
    __builtin_amdgcn_fence(__ATOMIC_ACQ_REL, "wavefront");
    __builtin_amdgcn_wave_barrier();

    // ---- stage 3: 8x WMMA (channel tiles) + depthwise reduction ---------
    // B layout: VGPR v holds packed K-pair {2v, 2v+1} (+8 if v>=4, +8 for lanes 16-31),
    // lane (l&15) = column N.  D: VGPR r = row M (kbase..kbase+7), lane = N.
    const int kbase = hiHalf * 8;
    float res[8];

    #pragma unroll
    for (int t = 0; t < 8; ++t) {
        const int c = t * 16 + M;
        v8u braw;
        #pragma unroll
        for (int v = 0; v < 8; ++v) {
            const int h2 = 2 * v + ((v & 4) ? 8 : 0) + (hiHalf ? 8 : 0);
            braw[v] = *(const unsigned int*)&XT[wave][c * XT_STRIDE + h2];
        }
        const v16h b = __builtin_bit_cast(v16h, braw);

        v8f acc = {};
        acc = __builtin_amdgcn_wmma_f32_16x16x32_f16(
            /*neg_a=*/false, a, /*neg_b=*/false, b,
            /*c_mod=*/(short)0, acc, /*reuse_a=*/false, /*reuse_b=*/false);

        // depthwise: lane's 8 weights are contiguous in dwT -> 2x ds_load_b128
        const float4 d0 = *(const float4*)&dwT[c * DWT_STRIDE + kbase + 0];
        const float4 d1 = *(const float4*)&dwT[c * DWT_STRIDE + kbase + 4];
        float partial = acc[0] * d0.x + acc[1] * d0.y + acc[2] * d0.z + acc[3] * d0.w
                      + acc[4] * d1.x + acc[5] * d1.y + acc[6] * d1.z + acc[7] * d1.w;
        // combine k=0..7 (lanes 0-15) with k=8..14 (lanes 16-31; k=15 weight 0)
        partial += __shfl_xor(partial, 16, 32);
        res[t] = partial;
    }

    // ---- stage 4: coalesced store (two tiles per 128B write) ------------
    #pragma unroll
    for (int i = 0; i < 4; ++i) {
        const float val = hiHalf ? res[2 * i + 1] : res[2 * i];
        const int   c   = i * 32 + lane;            // = (2i+hiHalf)*16 + M
        out[(long long)n * CH + c] = val + bias[c];
    }
}

extern "C" void kernel_launch(void* const* d_in, const int* in_sizes, int n_in,
                              void* d_out, int out_size, void* d_ws, size_t ws_size,
                              hipStream_t stream) {
    const float*     q_pts = (const float*)d_in[0];
    const float*     s_pts = (const float*)d_in[1];
    const long long* inds  = (const long long*)d_in[2];  // reference dtype is int64
    const float*     x     = (const float*)d_in[3];
    const float*     kp    = (const float*)d_in[4];
    const float*     dw    = (const float*)d_in[5];
    const float*     bias  = (const float*)d_in[6];
    float*           out   = (float*)d_out;

    const int blocks = NQ / WAVES_PER_BLOCK;   // 50000 / 4 = 12500, exact
    kpconv_wmma_kernel<<<blocks, THREADS, 0, stream>>>(
        q_pts, s_pts, inds, x, kp, dw, bias, out);
}
